// ResidualStack_81295140979149
// MI455X (gfx1250) — matrix operand order repaired and checked by reading, not compiled, not run on hardware
//
#include <hip/hip_runtime.h>
#include <hip/hip_bf16.h>

// ---------------------------------------------------------------------------
// WaveNet residual stack on gfx1250 (MI455X).
//   - all four GEMMs per block on v_wmma_f32_16x16x32_bf16
//   - residual stream kept in bf16, t-major [t][512]: x_next = 2*res is fully
//     recomputed per block and only consumed by the next block's bf16 GEMM,
//     so bf16 storage loses nothing vs converting at staging time.
//   - activation window staging via Tensor Data Mover (tensor_load_to_lds)
//     with D# built per the CDNA5 ISA bitfields; LDS row padding (8 halves)
//     done by the TDM pad feature; s_wait_tensorcnt + barrier to consume.
//   - weights converted once to bf16 (92 MB, resident in 192 MB L2).
// ---------------------------------------------------------------------------

typedef __attribute__((ext_vector_type(16))) __bf16   v16bf;
typedef __attribute__((ext_vector_type(8)))  float    v8f;
typedef __attribute__((ext_vector_type(4)))  unsigned u32x4;
typedef __attribute__((ext_vector_type(8)))  int      i32x8;
typedef __attribute__((ext_vector_type(4)))  int      i32x4;

#define RES_CHN 512
#define SKIP_CHN 256
#define NB 50
#define T_START 10000
#define SKIP_T 4885
#define NT 32          // timesteps per workgroup tile
#define ROW 520        // 512 + 8 halves pad (matches TDM pad_interval/amount)

#if defined(__HIP_DEVICE_COMPILE__) && __has_builtin(__builtin_amdgcn_tensor_load_to_lds)
#define HAVE_TDM 1
#else
#define HAVE_TDM 0
#endif

// ---------------- WMMA helper ----------------
__device__ __forceinline__ v8f wmma_bf16(v16bf a, v16bf b, v8f c) {
  // (neg_a, A, neg_b, B, c_mod, C, reuse_a, reuse_b)
  return __builtin_amdgcn_wmma_f32_16x16x32_bf16(false, a, false, b,
                                                 (short)0, c, false, false);
}

union Frag {
  uint4 q[2];
  v16bf v;
};

// A fragment (16x32, MxK) straight from row-major bf16 weights.
// lane 0-15: row M=lane, K={0..7,16..23}; lane 16-31: row M=lane-16,
// K={8..15,24..31} -> two contiguous 16B chunks per lane.
__device__ __forceinline__ v16bf load_a_frag(const __bf16* __restrict__ W,
                                             int mtile, int kbase, int lane) {
  int m  = mtile * 16 + (lane & 15);
  int kh = ((lane >> 4) & 1) * 8;
  const __bf16* p = W + (size_t)m * RES_CHN + kbase + kh;
  Frag f;
  f.q[0] = *(const uint4*)(p);
  f.q[1] = *(const uint4*)(p + 16);
  return f.v;
}

// B fragment (32x16, KxN) from LDS tile stored [t][c] (c contiguous).
// lane holds column N=lane&15; lanes 0-15 K={0..7,16..23}, 16-31
// K={8..15,24..31}: two contiguous 16B chunks per lane.
__device__ __forceinline__ v16bf load_b_frag(const __bf16* lds, int ntile,
                                             int kbase, int lane) {
  int n  = ntile * 16 + (lane & 15);
  int kh = ((lane >> 4) & 1) * 8;
  const __bf16* p = lds + n * ROW + kbase + kh;
  Frag f;
  f.q[0] = *(const uint4*)(p);
  f.q[1] = *(const uint4*)(p + 16);
  return f.v;
}

__device__ __forceinline__ float gated_act(float y) {
  float s = 1.0f / (1.0f + __expf(-y));
  return tanhf(y) * s;
}

// C/D layout: lane 0-15: (M=r, N=lane); lane 16-31: (M=r+8, N=lane-16).
__device__ __forceinline__ void store_gated(v8f acc, __bf16* gated, int mtile,
                                            int ntile, int lane) {
  int n  = ntile * 16 + (lane & 15);
  int cb = mtile * 16 + ((lane >> 4) & 1) * 8;
#pragma unroll
  for (int r = 0; r < 8; r += 2) {
    union { __bf16 h[2]; unsigned u; } pk;
    pk.h[0] = (__bf16)gated_act(acc[r]);
    pk.h[1] = (__bf16)gated_act(acc[r + 1]);
    *(unsigned*)(&gated[n * ROW + cb + r]) = pk.u;   // ds_store_b32
  }
}

// ---------------- TDM window load: bf16 [t][512] global -> [t][520] LDS ----
// D# per CDNA5 ISA:
//   group0: count=1 | lds_addr | global_addr(57b) | type=2
//   group1: data_size=2B, pad_enable, pad_interval=7 (256 DW = one 512-half
//           row), pad_amount=3 (4 DW = 8 halves)  -> LDS row pitch 520 halves
//           2-D tile: tile_dim0=512, tile_dim1=NT; tensor_dim1=rows_valid so
//           OOB rows zero-fill (replaces the tail guard).
__device__ __forceinline__ void tdm_load_window(const __bf16* gsrc,
                                                unsigned lds_byte,
                                                int rows_valid) {
#if HAVE_TDM
  unsigned long long ga = (unsigned long long)(size_t)gsrc;
  unsigned td1 = (unsigned)(rows_valid < 0 ? 0 : rows_valid);
  u32x4 g0;
  g0[0] = 1u;                                                // count=1
  g0[1] = lds_byte;                                          // lds_addr
  g0[2] = (unsigned)(ga & 0xFFFFFFFFu);                      // gaddr[31:0]
  g0[3] = (unsigned)((ga >> 32) & 0x1FFFFFFu) | (2u << 30);  // gaddr[56:32]|type=2
  i32x8 g1;
  g1[0] = (int)((1u << 16) | (1u << 20) | (7u << 22) | (3u << 25));
  g1[1] = (int)(((unsigned)RES_CHN & 0xFFFFu) << 16);        // tensor_dim0 lo
  g1[2] = (int)((td1 & 0xFFFFu) << 16);                      // {td1 lo, td0 hi}
  g1[3] = (int)(((unsigned)RES_CHN << 16) | ((td1 >> 16) & 0xFFFFu)); // {tile0, td1 hi}
  g1[4] = (int)NT;                                           // {tile2=0, tile1=NT}
  g1[5] = (int)RES_CHN;                                      // dim0_stride lo
  g1[6] = 0;
  g1[7] = 0;
  i32x4 z4 = {0, 0, 0, 0};
  i32x8 z8 = {0, 0, 0, 0, 0, 0, 0, 0};
  // 6-arg variant on this toolchain: (g0, g1, g2, g3, extra, cpol)
  __builtin_amdgcn_tensor_load_to_lds(g0, g1, z4, z4, z8, 0);
#else
  (void)gsrc; (void)lds_byte; (void)rows_valid;
#endif
}

// Manual fallback staging (also zero-fills OOB rows).
__device__ __forceinline__ void stage_window(const __bf16* __restrict__ src,
                                             int Tcur, int tbase, __bf16* dst,
                                             int tid) {
#pragma unroll 4
  for (int idx = tid; idx < NT * (RES_CHN / 8); idx += 256) {
    int t  = idx >> 6;     // 64 x 16B chunks per row
    int ch = idx & 63;
    uint4 v = {0u, 0u, 0u, 0u};
    if (tbase + t < Tcur)
      v = *(const uint4*)(src + (size_t)(tbase + t) * RES_CHN + ch * 8);
    *(uint4*)(dst + t * ROW + ch * 8) = v;
  }
}

// ---------------- one-time conversions ----------------
__global__ __launch_bounds__(256) void cvt_weights(
    const float* __restrict__ wdil, const float* __restrict__ wres,
    const float* __restrict__ wskip, __bf16* __restrict__ wdilb,
    __bf16* __restrict__ wresb, __bf16* __restrict__ wskipb) {
  const long NDIL = (long)NB * 2 * RES_CHN * RES_CHN;   // [blk][tap][o][c]
  const long NRES = (long)NB * RES_CHN * RES_CHN;
  const long NSK  = (long)NB * SKIP_CHN * RES_CHN;
  long total = NDIL + NRES + NSK;
  for (long idx = (long)blockIdx.x * blockDim.x + threadIdx.x; idx < total;
       idx += (long)gridDim.x * blockDim.x) {
    if (idx < NDIL) {
      long j = idx;
      int c = (int)(j & (RES_CHN - 1)); j >>= 9;
      int o = (int)(j & (RES_CHN - 1)); j >>= 9;
      int k = (int)(j & 1);             j >>= 1;
      int i = (int)j;
      wdilb[idx] = (__bf16)wdil[(((long)i * RES_CHN + o) * RES_CHN + c) * 2 + k];
    } else if (idx < NDIL + NRES) {
      long j = idx - NDIL;
      wresb[j] = (__bf16)wres[j];
    } else {
      long j = idx - NDIL - NRES;
      wskipb[j] = (__bf16)wskip[j];
    }
  }
}

// x: fp32 [C][T] -> bf16 [T][C] (one-time)
__global__ __launch_bounds__(256) void cvt_x(const float* __restrict__ x,
                                             __bf16* __restrict__ xb) {
  long total = (long)RES_CHN * T_START;
  for (long idx = (long)blockIdx.x * blockDim.x + threadIdx.x; idx < total;
       idx += (long)gridDim.x * blockDim.x) {
    int t = (int)(idx % T_START);   // t fastest -> coalesced reads
    int c = (int)(idx / T_START);
    xb[(size_t)t * RES_CHN + c] = (__bf16)x[idx];
  }
}

// ---------------- fused per-block kernel ----------------
__global__ __launch_bounds__(256) void wavenet_block(
    const __bf16* __restrict__ xin, __bf16* __restrict__ xout,
    const __bf16* __restrict__ w0, const __bf16* __restrict__ w1,
    const __bf16* __restrict__ wr, const __bf16* __restrict__ wsk,
    const float* __restrict__ br, const float* __restrict__ bs,
    float* __restrict__ skip_out, int Tcur, int Tnext, int dil) {
  extern __shared__ __bf16 smem[];
  __bf16* xa    = smem;                 // [NT][ROW] window at t0
  __bf16* xb    = smem + NT * ROW;      // [NT][ROW] window at t0+dil
  __bf16* gated = smem + 2 * NT * ROW;  // [NT][ROW] gated activations

  int tid  = threadIdx.x;
  int lane = tid & 31;
  int wave = tid >> 5;
  int t0   = blockIdx.x * NT;

#if HAVE_TDM
  if (wave == 0) {
    // flat shared-pointer low 32 bits == LDS byte offset (aperture mapping)
    unsigned lds_xa = (unsigned)(size_t)(void*)xa;
    unsigned lds_xb = (unsigned)(size_t)(void*)xb;
    tdm_load_window(xin + (size_t)t0 * RES_CHN, lds_xa, Tcur - t0);
    tdm_load_window(xin + (size_t)(t0 + dil) * RES_CHN, lds_xb,
                    Tcur - t0 - dil);
    __builtin_amdgcn_s_wait_tensorcnt(0);
  }
#else
  stage_window(xin, Tcur, t0, xa, tid);
  stage_window(xin, Tcur, t0 + dil, xb, tid);
#endif
  __syncthreads();

  // ---- stage 1: y = W0*xa + W1*xb, gate, -> LDS ----
  for (int mt = wave; mt < 32; mt += 8) {
    v8f acc0 = {}, acc1 = {};
#pragma unroll 4
    for (int ks = 0; ks < 16; ++ks) {   // K = 512 per tap
      int kb = ks * 32;
      v16bf a0  = load_a_frag(w0, mt, kb, lane);
      v16bf a1  = load_a_frag(w1, mt, kb, lane);
      v16bf b00 = load_b_frag(xa, 0, kb, lane);
      v16bf b01 = load_b_frag(xa, 1, kb, lane);
      v16bf b10 = load_b_frag(xb, 0, kb, lane);
      v16bf b11 = load_b_frag(xb, 1, kb, lane);
      acc0 = wmma_bf16(a0, b00, acc0);
      acc1 = wmma_bf16(a0, b01, acc1);
      acc0 = wmma_bf16(a1, b10, acc0);
      acc1 = wmma_bf16(a1, b11, acc1);
    }
    store_gated(acc0, gated, mt, 0, lane);
    store_gated(acc1, gated, mt, 1, lane);
  }
  __syncthreads();

  // ---- stage 2: res (M-tiles 0..31) and skip (M-tiles 32..47) ----
  for (int mt = wave; mt < 48; mt += 8) {
    bool isres = mt < 32;
    int mtl = isres ? mt : mt - 32;
    const __bf16* W = isres ? wr : wsk;
    v8f acc0 = {}, acc1 = {};
#pragma unroll 4
    for (int ks = 0; ks < 16; ++ks) {
      int kb = ks * 32;
      v16bf a  = load_a_frag(W, mtl, kb, lane);
      v16bf b0 = load_b_frag(gated, 0, kb, lane);
      v16bf b1 = load_b_frag(gated, 1, kb, lane);
      acc0 = wmma_bf16(a, b0, acc0);
      acc1 = wmma_bf16(a, b1, acc1);
    }
    int mb = mtl * 16 + ((lane >> 4) & 1) * 8;
    const float* bias = isres ? br : bs;
#pragma unroll
    for (int nt = 0; nt < 2; ++nt) {
      v8f acc = nt ? acc1 : acc0;
      int tn = t0 + nt * 16 + (lane & 15);
      if (tn >= Tnext) continue;
      if (isres) {
        // x_next = 2*res, packed bf16 t-major: one b128 store per lane
        union { __bf16 h[8]; uint4 q; } pk;
#pragma unroll
        for (int r = 0; r < 8; ++r) {
          float v = acc[r] + bias[mb + r];
          pk.h[r] = (__bf16)(v + v);
        }
        *(uint4*)(xout + (size_t)tn * RES_CHN + mb) = pk.q;
      } else {
        int s = tn - (Tnext - SKIP_T);
        if (s >= 0) {
#pragma unroll
          for (int r = 0; r < 8; ++r) {
            float v = acc[r] + bias[mb + r];
            skip_out[(size_t)(mb + r) * SKIP_T + s] = v;
          }
        }
      }
    }
  }
}

// ---------------- host launcher ----------------
extern "C" void kernel_launch(void* const* d_in, const int* in_sizes, int n_in,
                              void* d_out, int out_size, void* d_ws,
                              size_t ws_size, hipStream_t stream) {
  const float* x     = (const float*)d_in[0];
  const float* wdil  = (const float*)d_in[1];
  const float* wres  = (const float*)d_in[2];
  const float* bres  = (const float*)d_in[3];
  const float* wskip = (const float*)d_in[4];
  const float* bskip = (const float*)d_in[5];
  float* out = (float*)d_out;

  // workspace: bf16 weights (~92 MB) + bf16 x ping-pong (~20 MB)
  char* ws = (char*)d_ws;
  size_t off = 0;
  auto take = [&](size_t bytes) -> void* {
    void* p = ws + off;
    off += (bytes + 255) & ~(size_t)255;
    return p;
  };
  __bf16* wdilb  = (__bf16*)take((size_t)NB * 2 * RES_CHN * RES_CHN * 2);
  __bf16* wresb  = (__bf16*)take((size_t)NB * RES_CHN * RES_CHN * 2);
  __bf16* wskipb = (__bf16*)take((size_t)NB * SKIP_CHN * RES_CHN * 2);
  __bf16* xb0 = (__bf16*)take((size_t)T_START * RES_CHN * 2);
  __bf16* xb1 = (__bf16*)take((size_t)T_START * RES_CHN * 2);

  cvt_weights<<<2048, 256, 0, stream>>>(wdil, wres, wskip, wdilb, wresb, wskipb);
  cvt_x<<<1024, 256, 0, stream>>>(x, xb1);  // xb1 = initial x; overwritten
                                            // by block 1 after consumption

  int Tcur = T_START;
  const __bf16* xin = xb1;
  __bf16* bufs[2] = {xb0, xb1};
  for (int i = 0; i < NB; ++i) {
    int dil = 1 << (i % 10);
    int Tnext = Tcur - dil;
    __bf16* xout = bufs[i & 1];
    int nwg = (Tnext + NT - 1) / NT;
    size_t shmem = (size_t)3 * NT * ROW * sizeof(__bf16);  // 97.5 KB
    wavenet_block<<<nwg, 256, shmem, stream>>>(
        xin, xout,
        wdilb + (size_t)(i * 2 + 0) * RES_CHN * RES_CHN,
        wdilb + (size_t)(i * 2 + 1) * RES_CHN * RES_CHN,
        wresb + (size_t)i * RES_CHN * RES_CHN,
        wskipb + (size_t)i * SKIP_CHN * RES_CHN,
        bres + (size_t)i * RES_CHN, bskip + (size_t)i * SKIP_CHN,
        out + (size_t)i * SKIP_CHN * SKIP_T, Tcur, Tnext, dil);
    xin = xout;
    Tcur = Tnext;
  }
  (void)in_sizes; (void)n_in; (void)out_size; (void)ws_size;
}
